// AttentionMechanism_38474317037772
// MI455X (gfx1250) — matrix operand order repaired and checked by reading, not compiled, not run on hardware
//
#include <hip/hip_runtime.h>
#include <hip/hip_bf16.h>

// Problem constants (match the JAX reference)
#define B_   32
#define D_   1024   // CTX_DIM
#define C_   2048   // CTX_SIZE
#define H_   1024   // HID
#define DSLICES 4   // d-reduction split for the scores pass

typedef __attribute__((ext_vector_type(16))) __bf16 v16bf;
typedef __attribute__((ext_vector_type(8)))  float  v8f;

// ---------------------------------------------------------------------------
// Kernel 1: u = hidden @ W   (M=32, K=1024, N=1024), bf16 WMMA, f32 accumulate
// One wave per 16x16 output tile; 2 m-tiles * 64 n-tiles = 128 waves.
// A/B operand layouts follow CDNA5 ISA (16-bit A 16x32, B 32x16).
// ---------------------------------------------------------------------------
__global__ __launch_bounds__(256) void gemm_u_kernel(
    const float* __restrict__ hidden,  // [32, 1024] row-major
    const float* __restrict__ W,       // [1024(H), 1024(D)] row-major
    float* __restrict__ u)             // [32, 1024]
{
    const int lane  = threadIdx.x & 31;
    const int wib   = threadIdx.x >> 5;
    const int wave  = blockIdx.x * (blockDim.x >> 5) + wib;   // 0..127
    const int m0    = (wave & 1) * 16;                        // 0 or 16
    const int n0    = (wave >> 1) * 16;                       // 0..1008
    const bool hi   = lane >= 16;
    const int  l16  = lane & 15;

    v8f acc = {};
    for (int k0 = 0; k0 < H_; k0 += 32) {
        // A tile (16x32 bf16): lane<16 -> row m0+l16, K={k0+0..7, k0+16..23}
        //                      lane>=16 -> same rows,  K={k0+8..15, k0+24..31}
        v16bf a;
        {
            const float* arow = hidden + (size_t)(m0 + l16) * H_ + k0 + (hi ? 8 : 0);
            #pragma unroll
            for (int j = 0; j < 8; ++j) a[j]     = (__bf16)arow[j];
            #pragma unroll
            for (int j = 0; j < 8; ++j) a[8 + j] = (__bf16)arow[16 + j];
        }
        // B tile (32x16 bf16): lane<16 -> col n0+l16, K={k0+0..7, k0+16..23}
        //                      lane>=16 -> col n0+l16, K={k0+8..15, k0+24..31}
        v16bf bm;
        {
            const float* bcol = W + (size_t)(k0 + (hi ? 8 : 0)) * D_ + (n0 + l16);
            #pragma unroll
            for (int j = 0; j < 16; ++j) bm[j] = (__bf16)bcol[(size_t)(j < 8 ? j : j + 8) * D_];
        }
        acc = __builtin_amdgcn_wmma_f32_16x16x32_bf16(
            /*neg_a=*/false, a, /*neg_b=*/false, bm,
            /*c_mod=*/(short)0, acc, /*reuse_a=*/false, /*reuse_b=*/false);
    }
    // D layout: lanes 0-15: VGPR v -> M=m0+v; lanes 16-31: M=m0+v+8; N = n0+l16
    const int col = n0 + l16;
    #pragma unroll
    for (int v = 0; v < 8; ++v) {
        const int row = m0 + v + (hi ? 8 : 0);
        u[(size_t)row * D_ + col] = acc[v];
    }
}

// ---------------------------------------------------------------------------
// Kernel 2: partial scores, d-sliced:
//   spart[slice][b][c] = sum_{d in slice} ctxv[b,d,c] * u[b,d]
// float4 (b128) streaming over c; 256 workgroups keep all WGPs pulling HBM.
// grid = (C_/1024, B_, DSLICES), block = 256; each thread owns 4 columns.
// ---------------------------------------------------------------------------
__global__ __launch_bounds__(256) void scores_kernel(
    const float* __restrict__ ctxv,   // [B, D, C]
    const float* __restrict__ u,      // [B, D]
    float* __restrict__ spart)        // [DSLICES, B, C]
{
    __shared__ float us[D_ / DSLICES];               // 256 floats
    const int b     = blockIdx.y;
    const int slice = blockIdx.z;
    const int d0    = slice * (D_ / DSLICES);
    const int c4    = blockIdx.x * blockDim.x + threadIdx.x;   // float4 col idx

    us[threadIdx.x] = u[(size_t)b * D_ + d0 + threadIdx.x];
    __syncthreads();

    const float4* base =
        (const float4*)(ctxv + (size_t)b * D_ * C_ + (size_t)d0 * C_) + c4;
    float4 acc = {0.f, 0.f, 0.f, 0.f};
    for (int d = 0; d < D_ / DSLICES; d += 8) {
        if (d + 32 < D_ / DSLICES)
            __builtin_prefetch(base + (size_t)(d + 32) * (C_ / 4), 0, 1);
        #pragma unroll
        for (int j = 0; j < 8; ++j) {
            const float4 v = base[(size_t)(d + j) * (C_ / 4)];
            const float  w = us[d + j];
            acc.x = fmaf(v.x, w, acc.x);
            acc.y = fmaf(v.y, w, acc.y);
            acc.z = fmaf(v.z, w, acc.z);
            acc.w = fmaf(v.w, w, acc.w);
        }
    }
    float4* out = (float4*)(spart + ((size_t)slice * B_ + b) * C_) + c4;
    *out = acc;
}

// ---------------------------------------------------------------------------
// Kernel 3: fold DSLICES partials + softmax over c for each b.
// One block (256 thr = 8 waves) per b. Bias term dropped (softmax-invariant).
// ---------------------------------------------------------------------------
__global__ __launch_bounds__(256) void softmax_kernel(
    const float* __restrict__ spart,   // [DSLICES, B, C]
    float* __restrict__ attn)          // [B, C]
{
    const int b   = blockIdx.x;
    const int tid = threadIdx.x;
    __shared__ float red[8];
    __shared__ float stat[2];

    float v[C_ / 256];
    #pragma unroll
    for (int i = 0; i < C_ / 256; ++i) {
        const size_t idx = (size_t)b * C_ + tid + i * 256;
        float s = spart[idx];
        #pragma unroll
        for (int sl = 1; sl < DSLICES; ++sl)
            s += spart[(size_t)sl * B_ * C_ + idx];
        v[i] = s;
    }

    // block max
    float m = v[0];
    #pragma unroll
    for (int i = 1; i < C_ / 256; ++i) m = fmaxf(m, v[i]);
    #pragma unroll
    for (int off = 16; off > 0; off >>= 1) m = fmaxf(m, __shfl_xor(m, off, 32));
    if ((tid & 31) == 0) red[tid >> 5] = m;
    __syncthreads();
    if (tid == 0) {
        float mm = red[0];
        #pragma unroll
        for (int i = 1; i < 8; ++i) mm = fmaxf(mm, red[i]);
        stat[0] = mm;
    }
    __syncthreads();
    const float mm = stat[0];

    // exp + block sum
    float s = 0.f;
    #pragma unroll
    for (int i = 0; i < C_ / 256; ++i) { v[i] = __expf(v[i] - mm); s += v[i]; }
    #pragma unroll
    for (int off = 16; off > 0; off >>= 1) s += __shfl_xor(s, off, 32);
    if ((tid & 31) == 0) red[tid >> 5] = s;
    __syncthreads();
    if (tid == 0) {
        float ss = 0.f;
        #pragma unroll
        for (int i = 0; i < 8; ++i) ss += red[i];
        stat[1] = 1.0f / ss;
    }
    __syncthreads();
    const float inv = stat[1];
    #pragma unroll
    for (int i = 0; i < C_ / 256; ++i)
        attn[(size_t)b * C_ + tid + i * 256] = v[i] * inv;
}

// ---------------------------------------------------------------------------
// Kernel 4: ctx[b,d] = sum_c ctxv[b,d,c] * attn[b,c]
// One wave per d-row, float4 streaming (16 b128 loads per wave);
// attn row staged in LDS.  grid = (D_/8, B_), block = 256 (8 waves)
// ---------------------------------------------------------------------------
__global__ __launch_bounds__(256) void ctx_kernel(
    const float* __restrict__ ctxv,   // [B, D, C]
    const float* __restrict__ attn,   // [B, C]
    float* __restrict__ ctxo)         // [B, D]
{
    __shared__ float4 as4[C_ / 4];    // 8 KB
    const int b    = blockIdx.y;
    const int tid  = threadIdx.x;
    const int wave = tid >> 5;
    const int lane = tid & 31;
    const float4* arow = (const float4*)(attn + (size_t)b * C_);
    for (int i = tid; i < C_ / 4; i += blockDim.x) as4[i] = arow[i];
    __syncthreads();

    const int d = blockIdx.x * 8 + wave;
    const float4* row = (const float4*)(ctxv + (size_t)b * D_ * C_ + (size_t)d * C_);
    float acc = 0.f;
    #pragma unroll 4
    for (int i = lane; i < C_ / 4; i += 32) {
        const float4 v = row[i];
        const float4 w = as4[i];
        acc = fmaf(v.x, w.x, fmaf(v.y, w.y, fmaf(v.z, w.z, fmaf(v.w, w.w, acc))));
    }
    #pragma unroll
    for (int off = 16; off > 0; off >>= 1) acc += __shfl_xor(acc, off, 32);
    if (lane == 0) ctxo[(size_t)b * D_ + d] = acc;
}

// ---------------------------------------------------------------------------
// Kernel 5: out[s,b,d] = ctx[b,d], s = 0..seqlen-1. float4 coalesced stores.
// ---------------------------------------------------------------------------
__global__ __launch_bounds__(256) void bcast_kernel(
    const float* __restrict__ ctxo,   // [B, D]
    float* __restrict__ out,          // [seqlen, B, D]
    int seqlen)
{
    const int t = blockIdx.x * blockDim.x + threadIdx.x;  // float4 idx, 0..8191
    const float4 v = ((const float4*)ctxo)[t];
    float4* o = (float4*)out;
    for (int s = 0; s < seqlen; ++s)
        o[(size_t)s * (B_ * D_ / 4) + t] = v;
}

// ---------------------------------------------------------------------------
extern "C" void kernel_launch(void* const* d_in, const int* in_sizes, int n_in,
                              void* d_out, int out_size, void* d_ws, size_t ws_size,
                              hipStream_t stream) {
    // setup_inputs order: seqlen, hidden, contextvects, W, b
    const float* hidden = (const float*)d_in[1];   // [1,32,1024]
    const float* ctxv   = (const float*)d_in[2];   // [32,1024,2048]
    const float* W      = (const float*)d_in[3];   // [1024,1024]
    // d_in[4] (bias) intentionally unused: per-row constant, softmax-invariant.

    const int seqlen = out_size / (B_ * D_);       // = 64 (avoid device read of scalar)

    float* ws     = (float*)d_ws;
    float* u      = ws;                             // 32*1024
    float* spart  = u + B_ * D_;                    // DSLICES*32*2048
    float* attn   = spart + DSLICES * B_ * C_;      // 32*2048
    float* ctxo   = attn + B_ * C_;                 // 32*1024

    // 1) u = hidden @ W  (WMMA bf16): 128 waves = 16 blocks * 8 waves
    gemm_u_kernel<<<dim3(16), dim3(256), 0, stream>>>(hidden, W, u);
    // 2) d-sliced partial scores (streams 256 MB of ctxv, b128 loads)
    scores_kernel<<<dim3(C_ / 1024, B_, DSLICES), dim3(256), 0, stream>>>(ctxv, u, spart);
    // 3) fold partials + softmax per batch row
    softmax_kernel<<<dim3(B_), dim3(256), 0, stream>>>(spart, attn);
    // 4) attention-weighted context (streams 256 MB of ctxv, b128 loads)
    ctx_kernel<<<dim3(D_ / 8, B_), dim3(256), 0, stream>>>(ctxv, attn, ctxo);
    // 5) broadcast across seqlen
    bcast_kernel<<<dim3((B_ * D_ / 4) / 256), dim3(256), 0, stream>>>(ctxo, (float*)d_out, seqlen);
}
